// TransformerBlock_48112223650153
// MI455X (gfx1250) — matrix operand order repaired and checked by reading, not compiled, not run on hardware
//
#include <hip/hip_runtime.h>
#include <hip/hip_bf16.h>

typedef __bf16 bf16_t;
typedef __bf16 bf16x16 __attribute__((ext_vector_type(16)));
typedef float  f32x8   __attribute__((ext_vector_type(8)));
typedef int    v4i_t   __attribute__((ext_vector_type(4)));

static __device__ __forceinline__ f32x8 wmma_bf16(bf16x16 a, bf16x16 b, f32x8 c) {
    // D = A(16x32 bf16) * B(32x16 bf16) + C(16x16 f32)
    return __builtin_amdgcn_wmma_f32_16x16x32_bf16(
        /*neg_a=*/false, a, /*neg_b=*/false, b,
        /*c_mod=*/(short)0, c, /*reuse_a=*/false, /*reuse_b=*/false);
}

// 16-byte global -> LDS copy; async-tensor path on gfx1250 if available.
static __device__ __forceinline__ void async_copy16(const bf16_t* gsrc, bf16_t* ldst) {
#if defined(__gfx1250__) && __has_builtin(__builtin_amdgcn_global_load_async_to_lds_b128)
    __builtin_amdgcn_global_load_async_to_lds_b128(
        (__attribute__((address_space(1))) v4i_t*)(gsrc),
        (__attribute__((address_space(3))) v4i_t*)(ldst), 0, 0);
#else
    *reinterpret_cast<uint4*>(ldst) = *reinterpret_cast<const uint4*>(gsrc);
#endif
}

// Wait until at most N async LDS loads remain outstanding (in-order completion,
// so N = size of the next pipeline stage keeps it in flight).
template <int N>
static __device__ __forceinline__ void async_wait() {
#if defined(__gfx1250__) && __has_builtin(__builtin_amdgcn_global_load_async_to_lds_b128)
#if __has_builtin(__builtin_amdgcn_s_wait_asynccnt)
    __builtin_amdgcn_s_wait_asynccnt(N);
#else
    asm volatile("s_wait_asynccnt %0" :: "i"(N) : "memory");
#endif
#endif
}

// ---------------------------------------------------------------------------
// f32 -> bf16 conversion (row-major copy)
// ---------------------------------------------------------------------------
__global__ __launch_bounds__(256) void cvt_bf16_kernel(const float* __restrict__ in,
                                                       bf16_t* __restrict__ out, int n) {
    int i = blockIdx.x * 256 + threadIdx.x;
    if (i < n) out[i] = (bf16_t)in[i];
}

// ---------------------------------------------------------------------------
// f32 [K,N] -> bf16 [N,K] transposed convert (32x32 LDS tiles, 256 threads)
// ---------------------------------------------------------------------------
__global__ __launch_bounds__(256) void cvt_transpose_kernel(const float* __restrict__ in,
                                                            bf16_t* __restrict__ out,
                                                            int K, int N) {
    __shared__ float tile[32][33];
    const int k0 = blockIdx.y * 32, n0 = blockIdx.x * 32;
    const int tx = threadIdx.x & 31, ty = threadIdx.x >> 5;  // 32 x 8
    #pragma unroll
    for (int j = 0; j < 32; j += 8)
        tile[ty + j][tx] = in[(size_t)(k0 + ty + j) * N + n0 + tx];
    __syncthreads();
    #pragma unroll
    for (int j = 0; j < 32; j += 8)
        out[(size_t)(n0 + ty + j) * K + k0 + tx] = (bf16_t)tile[tx][ty + j];
}

// ---------------------------------------------------------------------------
// Tiled WMMA GEMM: C[M,N] = A[M,K](bf16, row-major) @ W^T  (WT is [N,K] bf16)
// Block tile 128x128, BK=32, 256 threads = 8 waves arranged 4(M) x 2(N),
// each wave computes a 32x64 tile = 8 WMMAs per K-step with fragment reuse.
// Double-buffered LDS with async-to-LDS staging pipelined one K-step ahead.
// MODE 0: store f32. MODE 1: GELU -> bf16. MODE 2: QKV split -> q/k/v bf16
//         (q,k stored [b,h,n,dh]; v stored transposed [b,h,dh,n]).
// ---------------------------------------------------------------------------
template <int MODE>
__global__ __launch_bounds__(256) void gemm_bf16_kernel(
    const bf16_t* __restrict__ A, const bf16_t* __restrict__ WT,
    const float* __restrict__ bias,
    float* __restrict__ Cf,
    bf16_t* __restrict__ O0, bf16_t* __restrict__ O1, bf16_t* __restrict__ O2,
    int M, int N, int K)
{
    __shared__ bf16_t As[2][128 * 32];
    __shared__ bf16_t Bs[2][128 * 32];   // rows = N-columns (from WT), cols = K

    const int tid  = threadIdx.x;
    const int lane = tid & 31;
    const int wave = tid >> 5;
    const int half = (lane >> 4) & 1;
    const int lr   = lane & 15;
    const int wm   = wave >> 1;   // 0..3  (M sub-block of 32 rows)
    const int wn   = wave & 1;    // 0..1  (N sub-block of 64 cols)
    const int bm   = blockIdx.y * 128;
    const int bn   = blockIdx.x * 128;

    f32x8 acc[2][4];
    #pragma unroll
    for (int mi = 0; mi < 2; ++mi)
        #pragma unroll
        for (int ni = 0; ni < 4; ++ni)
            #pragma unroll
            for (int v = 0; v < 8; ++v) acc[mi][ni][v] = 0.0f;

    const int sr = tid >> 1;            // 0..127 staging row
    const int sc = (tid & 1) * 16;      // 0 or 16

    // Issue a K-step's tile loads (4 x b128 async per thread) into buffer `buf`.
    auto stage = [&](int buf, int k0) {
        async_copy16(&A [(size_t)(bm + sr) * K + k0 + sc],     &As[buf][sr * 32 + sc]);
        async_copy16(&A [(size_t)(bm + sr) * K + k0 + sc + 8], &As[buf][sr * 32 + sc + 8]);
        async_copy16(&WT[(size_t)(bn + sr) * K + k0 + sc],     &Bs[buf][sr * 32 + sc]);
        async_copy16(&WT[(size_t)(bn + sr) * K + k0 + sc + 8], &Bs[buf][sr * 32 + sc + 8]);
    };

    const int T = K >> 5;
    stage(0, 0);
    for (int kt = 0; kt < T; ++kt) {
        const int cur = kt & 1;
        if (kt + 1 < T) { stage(cur ^ 1, (kt + 1) << 5); async_wait<4>(); }
        else            { async_wait<0>(); }
        __syncthreads();   // current buffer visible to all waves

        // A fragments: a[e] = As[row][k], k(e) = e + 16*half (contiguous)
        union { bf16x16 v; uint4 u[2]; } af[2];
        #pragma unroll
        for (int mi = 0; mi < 2; ++mi) {
            const bf16_t* arow = &As[cur][(wm * 32 + mi * 16 + lr) * 32];
            af[mi].u[0] = *reinterpret_cast<const uint4*>(arow + 16 * half);
            af[mi].u[1] = *reinterpret_cast<const uint4*>(arow + 16 * half + 8);
        }
        // B fragments: b[e] = B[k=e+16*half][n] = Bs[n][e+16*half] (contiguous)
        union { bf16x16 v; uint4 u[2]; } bfr[4];
        #pragma unroll
        for (int ni = 0; ni < 4; ++ni) {
            const bf16_t* brow = &Bs[cur][(wn * 64 + ni * 16 + lr) * 32 + 16 * half];
            bfr[ni].u[0] = *reinterpret_cast<const uint4*>(brow);
            bfr[ni].u[1] = *reinterpret_cast<const uint4*>(brow + 8);
        }
        #pragma unroll
        for (int mi = 0; mi < 2; ++mi)
            #pragma unroll
            for (int ni = 0; ni < 4; ++ni)
                acc[mi][ni] = wmma_bf16(af[mi].v, bfr[ni].v, acc[mi][ni]);
        __syncthreads();   // release `cur` before it is re-staged two steps later
    }

    // Epilogue: fragment element v -> row M = v + 8*half, col N = lane%16
    #pragma unroll
    for (int mi = 0; mi < 2; ++mi) {
        #pragma unroll
        for (int ni = 0; ni < 4; ++ni) {
            #pragma unroll
            for (int v = 0; v < 8; ++v) {
                const int gr = bm + wm * 32 + mi * 16 + v + 8 * half;
                const int gc = bn + wn * 64 + ni * 16 + lr;
                float val = acc[mi][ni][v] + bias[gc];
                if (MODE == 0) {
                    Cf[(size_t)gr * N + gc] = val;
                } else if (MODE == 1) {
                    float ge = 0.5f * val * (1.0f + erff(val * 0.70710678118654752f));
                    O0[(size_t)gr * N + gc] = (bf16_t)ge;
                } else {
                    // col c = h*192 + dh*3 + s  (qkv innermost per reference reshape)
                    int h = gc / 192, rem = gc % 192, dh = rem / 3, s = rem % 3;
                    int bb = gr >> 10, n = gr & 1023;
                    if (s == 2) {  // V stored transposed: [b, h, dh, n]
                        O2[((((size_t)bb * 12 + h) * 64 + dh) << 10) + n] = (bf16_t)val;
                    } else {
                        bf16_t* dst = (s == 0) ? O0 : O1;
                        dst[((((size_t)bb * 12 + h) << 10) + n) * 64 + dh] = (bf16_t)val;
                    }
                }
            }
        }
    }
}

// ---------------------------------------------------------------------------
// Flash attention, bf16 WMMA, dh=64, seq=1024, post-softmax scale 1/sqrt(768).
// Q/K layout [b,h,n,64]; V layout [b,h,64,n] (transposed). Grid (b*h, 1024/128).
// Each wave owns a 16-row Q tile; K/V double-buffered in LDS in 32-key blocks
// with async staging pipelined one block ahead.
// ---------------------------------------------------------------------------
__global__ __launch_bounds__(256) void attention_bf16_kernel(
    const bf16_t* __restrict__ Q, const bf16_t* __restrict__ Km,
    const bf16_t* __restrict__ V, bf16_t* __restrict__ O)
{
    __shared__ bf16_t Ks[2][32 * 64];    // [key][dh]
    __shared__ bf16_t Vs[2][64 * 32];    // [dh][key]  (from transposed V)
    __shared__ bf16_t Ps[8 * 16 * 32];

    const int tid  = threadIdx.x;
    const int lane = tid & 31;
    const int wave = tid >> 5;
    const int half = (lane >> 4) & 1;
    const int lr   = lane & 15;
    const int bh   = blockIdx.x;          // b*12 + h
    const int b    = bh / 12, h = bh % 12;
    const int q0   = blockIdx.y * 128 + wave * 16;

    // Q fragments (two K-chunks of 32 over dh=64)
    union { bf16x16 v; uint4 u[2]; } qa[2];
    const bf16_t* qrow = Q + (((size_t)bh << 10) + q0 + lr) * 64;
    #pragma unroll
    for (int c = 0; c < 2; ++c) {
        qa[c].u[0] = *reinterpret_cast<const uint4*>(qrow + c * 32 + 16 * half);
        qa[c].u[1] = *reinterpret_cast<const uint4*>(qrow + c * 32 + 16 * half + 8);
    }

    f32x8 o_acc[4];
    #pragma unroll
    for (int i = 0; i < 4; ++i)
        #pragma unroll
        for (int v = 0; v < 8; ++v) o_acc[i][v] = 0.0f;
    float m_r[8], l_r[8];
    #pragma unroll
    for (int v = 0; v < 8; ++v) { m_r[v] = -1e30f; l_r[v] = 0.0f; }

    const bf16_t* Kbase = Km + (((size_t)bh) << 10) * 64;   // [n][dh]
    const bf16_t* Vbase = V + (((size_t)bh * 64) << 10);    // [dh][n]

    // Stage one 32-key block (2 x b128 async per thread) into buffer `buf`.
    auto stage = [&](int buf, int kb) {
        int r = tid >> 3, c = (tid & 7) * 8;             // Ks: 32x64
        async_copy16(&Kbase[(size_t)(kb + r) * 64 + c], &Ks[buf][r * 64 + c]);
        int r2 = tid >> 2, c2 = (tid & 3) * 8;           // Vs: 64x32
        async_copy16(&Vbase[((size_t)r2 << 10) + kb + c2], &Vs[buf][r2 * 32 + c2]);
    };

    stage(0, 0);
    for (int it = 0; it < 32; ++it) {
        const int cur = it & 1;
        if (it + 1 < 32) { stage(cur ^ 1, (it + 1) * 32); async_wait<2>(); }
        else             { async_wait<0>(); }
        __syncthreads();

        // S[16,32] = Q[16,64] @ K_blk[32,64]^T
        f32x8 s[2];
        #pragma unroll
        for (int nt = 0; nt < 2; ++nt)
            #pragma unroll
            for (int v = 0; v < 8; ++v) s[nt][v] = 0.0f;
        #pragma unroll
        for (int c2 = 0; c2 < 2; ++c2) {
            #pragma unroll
            for (int nt = 0; nt < 2; ++nt) {
                union { bf16x16 v; uint4 u[2]; } kf;
                const bf16_t* krow = &Ks[cur][(nt * 16 + lr) * 64 + c2 * 32 + 16 * half];
                kf.u[0] = *reinterpret_cast<const uint4*>(krow);
                kf.u[1] = *reinterpret_cast<const uint4*>(krow + 8);
                s[nt] = wmma_bf16(qa[c2].v, kf.v, s[nt]);
            }
        }

        // Online softmax (row reductions across the 16-lane half-wave)
        #pragma unroll
        for (int v = 0; v < 8; ++v) {
            float t = fmaxf(s[0][v], s[1][v]);
            #pragma unroll
            for (int off = 1; off < 16; off <<= 1) t = fmaxf(t, __shfl_xor(t, off, 16));
            float mn = fmaxf(m_r[v], t);
            float sc = __expf(m_r[v] - mn);
            float p0 = __expf(s[0][v] - mn);
            float p1 = __expf(s[1][v] - mn);
            float rs = p0 + p1;
            #pragma unroll
            for (int off = 1; off < 16; off <<= 1) rs += __shfl_xor(rs, off, 16);
            l_r[v] = l_r[v] * sc + rs;
            m_r[v] = mn;
            #pragma unroll
            for (int vt = 0; vt < 4; ++vt) o_acc[vt][v] *= sc;
            bf16_t* prow = &Ps[(wave * 16 + v + 8 * half) * 32];
            prow[lr]      = (bf16_t)p0;
            prow[16 + lr] = (bf16_t)p1;
        }
        asm volatile("" ::: "memory");  // keep P stores before P loads (same-wave LDS)

        // O[16,64] += P[16,32] @ V_blk[32,64]
        union { bf16x16 v; uint4 u[2]; } pf;
        const bf16_t* pr = &Ps[(wave * 16 + lr) * 32];
        pf.u[0] = *reinterpret_cast<const uint4*>(pr + 16 * half);
        pf.u[1] = *reinterpret_cast<const uint4*>(pr + 16 * half + 8);
        #pragma unroll
        for (int vt = 0; vt < 4; ++vt) {
            union { bf16x16 v; uint4 u[2]; } vf;  // b[e] = Vs[dh=vt*16+lr][k=e+16*half]
            const bf16_t* vrow = &Vs[cur][(vt * 16 + lr) * 32 + 16 * half];
            vf.u[0] = *reinterpret_cast<const uint4*>(vrow);
            vf.u[1] = *reinterpret_cast<const uint4*>(vrow + 8);
            o_acc[vt] = wmma_bf16(pf.v, vf.v, o_acc[vt]);
        }
        __syncthreads();   // release `cur` before it is re-staged two steps later
    }

    // Normalize (1/l) and apply the reference's post-softmax 1/sqrt(768) scale.
    const float s768 = 0.03608439182435161f;
    #pragma unroll
    for (int vt = 0; vt < 4; ++vt) {
        #pragma unroll
        for (int v = 0; v < 8; ++v) {
            float val = o_acc[vt][v] / l_r[v] * s768;
            int gq = q0 + v + 8 * half;
            int gc = h * 64 + vt * 16 + lr;
            O[(((size_t)b << 10) + gq) * 768 + gc] = (bf16_t)val;
        }
    }
}

// ---------------------------------------------------------------------------
// out = x + LayerNorm(y)*g + b ; optionally also bf16 copy of out.
// One row (768) per 256-thread block.
// ---------------------------------------------------------------------------
__global__ __launch_bounds__(256) void residual_ln_kernel(
    const float* __restrict__ x, const float* __restrict__ y,
    const float* __restrict__ g, const float* __restrict__ be,
    float* __restrict__ out, bf16_t* __restrict__ outb)
{
    __shared__ float r1[256], r2[256];
    const int row = blockIdx.x, tid = threadIdx.x;
    const float* yr = y + (size_t)row * 768;
    float v0 = yr[tid], v1 = yr[tid + 256], v2 = yr[tid + 512];
    r1[tid] = v0 + v1 + v2;
    r2[tid] = v0 * v0 + v1 * v1 + v2 * v2;
    __syncthreads();
    for (int s = 128; s > 0; s >>= 1) {
        if (tid < s) { r1[tid] += r1[tid + s]; r2[tid] += r2[tid + s]; }
        __syncthreads();
    }
    const float mu  = r1[0] * (1.0f / 768.0f);
    const float var = r2[0] * (1.0f / 768.0f) - mu * mu;
    const float rs  = rsqrtf(var + 1e-5f);
    const float* xr = x + (size_t)row * 768;
    float* orow = out + (size_t)row * 768;
    #pragma unroll
    for (int j = 0; j < 3; ++j) {
        int i = tid + j * 256;
        float val = xr[i] + (yr[i] - mu) * rs * g[i] + be[i];
        orow[i] = val;
        if (outb) outb[(size_t)row * 768 + i] = (bf16_t)val;
    }
}

// ---------------------------------------------------------------------------
// Host launch
// ---------------------------------------------------------------------------
extern "C" void kernel_launch(void* const* d_in, const int* in_sizes, int n_in,
                              void* d_out, int out_size, void* d_ws, size_t ws_size,
                              hipStream_t stream) {
    const int B = 8, N = 1024, D = 768, H = 12, HID = 3072;
    const int M = B * N;  // 8192 rows

    const float* x     = (const float*)d_in[0];
    const float* Wqkv  = (const float*)d_in[1];
    const float* bqkv  = (const float*)d_in[2];
    const float* Wproj = (const float*)d_in[3];
    const float* bproj = (const float*)d_in[4];
    const float* W1    = (const float*)d_in[5];
    const float* b1    = (const float*)d_in[6];
    const float* W2    = (const float*)d_in[7];
    const float* b2    = (const float*)d_in[8];
    const float* g1    = (const float*)d_in[9];
    const float* be1   = (const float*)d_in[10];
    const float* g2    = (const float*)d_in[11];
    const float* be2   = (const float*)d_in[12];
    float* out = (float*)d_out;

    char* base = (char*)d_ws;
    size_t off = 0;
    auto alloc = [&](size_t bytes) -> char* {
        off = (off + 255) & ~(size_t)255;
        char* p = base + off;
        off += bytes;
        return p;
    };
    bf16_t* xb      = (bf16_t*)alloc((size_t)M * D * 2);
    bf16_t* wqkv_t  = (bf16_t*)alloc((size_t)D * 3 * D * 2);   // [3D, D]
    bf16_t* wproj_t = (bf16_t*)alloc((size_t)D * D * 2);       // [D, D]
    bf16_t* w1_t    = (bf16_t*)alloc((size_t)D * HID * 2);     // [HID, D]
    bf16_t* w2_t    = (bf16_t*)alloc((size_t)HID * D * 2);     // [D, HID]
    bf16_t* qb      = (bf16_t*)alloc((size_t)M * D * 2);       // [b,h,n,dh]
    bf16_t* kb      = (bf16_t*)alloc((size_t)M * D * 2);       // [b,h,n,dh]
    bf16_t* vb      = (bf16_t*)alloc((size_t)M * D * 2);       // [b,h,dh,n]
    bf16_t* ob      = (bf16_t*)alloc((size_t)M * D * 2);
    float*  branch  = (float*)alloc((size_t)M * D * 4);        // proj out / MLP out
    float*  x1      = (float*)alloc((size_t)M * D * 4);
    bf16_t* x1b     = (bf16_t*)alloc((size_t)M * D * 2);
    bf16_t* h1      = (bf16_t*)alloc((size_t)M * HID * 2);
    (void)ws_size; (void)n_in; (void)in_sizes; (void)out_size;

    // Convert activations (row-major) and weights (transposed to [N,K]).
    cvt_bf16_kernel<<<(M * D + 255) / 256, 256, 0, stream>>>(x, xb, M * D);
    cvt_transpose_kernel<<<dim3((3 * D) / 32, D / 32), 256, 0, stream>>>(Wqkv, wqkv_t, D, 3 * D);
    cvt_transpose_kernel<<<dim3(D / 32, D / 32), 256, 0, stream>>>(Wproj, wproj_t, D, D);
    cvt_transpose_kernel<<<dim3(HID / 32, D / 32), 256, 0, stream>>>(W1, w1_t, D, HID);
    cvt_transpose_kernel<<<dim3(D / 32, HID / 32), 256, 0, stream>>>(W2, w2_t, HID, D);

    // QKV projection with de-interleaving epilogue (V transposed).
    gemm_bf16_kernel<2><<<dim3((3 * D) / 128, M / 128), 256, 0, stream>>>(
        xb, wqkv_t, bqkv, nullptr, qb, kb, vb, M, 3 * D, D);

    // Attention
    attention_bf16_kernel<<<dim3(B * H, N / 128), 256, 0, stream>>>(qb, kb, vb, ob);

    // Output projection
    gemm_bf16_kernel<0><<<dim3(D / 128, M / 128), 256, 0, stream>>>(
        ob, wproj_t, bproj, branch, nullptr, nullptr, nullptr, M, D, D);

    // x1 = x + LN(attn_out)
    residual_ln_kernel<<<M, 256, 0, stream>>>(x, branch, g1, be1, x1, x1b);

    // MLP up + GELU
    gemm_bf16_kernel<1><<<dim3(HID / 128, M / 128), 256, 0, stream>>>(
        x1b, w1_t, b1, nullptr, h1, nullptr, nullptr, M, HID, D);

    // MLP down
    gemm_bf16_kernel<0><<<dim3(D / 128, M / 128), 256, 0, stream>>>(
        h1, w2_t, b2, branch, nullptr, nullptr, nullptr, M, D, HID);

    // out = x1 + LN(mlp_out)
    residual_ln_kernel<<<M, 256, 0, stream>>>(x1, branch, g2, be2, out, nullptr);
}
